// QRNN3DLayer_14946486190563
// MI455X (gfx1250) — matrix operand order repaired and checked by reading, not compiled, not run on hardware
//
#include <hip/hip_runtime.h>
#include <stdint.h>

#define B_   4
#define CIN  16
#define T_   31
#define H_   128
#define W_   128
#define HID  16

typedef __attribute__((ext_vector_type(16))) __bf16 v16bf;
typedef __attribute__((ext_vector_type(8)))  float  v8f;
typedef __attribute__((ext_vector_type(8)))  unsigned short ushort8;

union Frag { v16bf v; ushort8 h[2]; };

__device__ __forceinline__ unsigned short f2bf(float f) {
  union { float f; unsigned u; } x; x.f = f;
  unsigned r = (x.u + 0x7FFFu + ((x.u >> 16) & 1u)) >> 16;
  return (unsigned short)r;
}

// sigmoid / tanh via hardware exp (v_exp_f32); saturate correctly at +/-inf
__device__ __forceinline__ float sigm(float x) { return 1.0f / (1.0f + __expf(-x)); }
__device__ __forceinline__ float tanh_f(float x) { return 1.0f - 2.0f / (__expf(2.0f * x) + 1.0f); }

static constexpr int TILE  = 16;                  // spatial tile edge
static constexpr int HALO  = 18;                  // TILE + 2
static constexpr int SH    = HALO * HALO * CIN;   // 5184 halves per time slice
static constexpr int WK    = 448;                 // K padded (432 -> 448) = 14 chunks of 32
static constexpr int NFULL = SH / 256;            // 20 full rounds per thread
static constexpr int NTAIL = SH % 256;            // 64-thread tail

__global__ __launch_bounds__(256)
void qrnn3d_fused(const float* __restrict__ x, const float* __restrict__ Wt,
                  const float* __restrict__ bias, float* __restrict__ out)
{
  __shared__ __align__(16) unsigned short xs[4][SH];        // bf16 x ring buffer, 4 deep
  __shared__ __align__(16) unsigned short wlds[32 * WK];    // bf16 weights [o][k], k = ci + 16*tap

  const int tid  = threadIdx.x;
  const int lane = tid & 31;
  const int wave = tid >> 5;            // 0..7, each wave owns 2 M-tiles (2 h-rows)
  const int blk  = blockIdx.x;          // 256 blocks = 4 batches * 64 tiles
  const int b    = blk >> 6;
  const int tl   = blk & 63;
  const int h0   = (tl >> 3) * TILE;
  const int w0   = (tl & 7) * TILE;

  // ---- stage weights: W[o][ci][kd][kh][kw] -> wlds[o][ci + 16*tap], tap = kd*9+kh*3+kw
  for (int idx = tid; idx < 32 * 432; idx += 256) {
    int o = idx / 432, rem = idx % 432;
    int ci = rem / 27, tap = rem % 27;
    wlds[o * WK + tap * 16 + ci] = f2bf(Wt[idx]);
  }
  for (int idx = tid; idx < 32 * 16; idx += 256)            // zero K-pad 432..447
    wlds[(idx >> 4) * WK + 432 + (idx & 15)] = 0;

  // ---- hoisted per-thread slice-copy offsets (t-invariant)
  int gofs[NFULL + 1];   // global BYTE offset at tt=0, or -1 for zero-fill
  int lofs[NFULL + 1];   // LDS half offset within a slice
  float vals[NFULL + 1];
#pragma unroll
  for (int i = 0; i <= NFULL; ++i) {
    int idx = tid + i * 256;
    int ci = idx / (HALO * HALO), rem = idx % (HALO * HALO);
    int hh = rem / HALO, ww = rem % HALO;
    int hg = h0 + hh - 1, wg = w0 + ww - 1;
    lofs[i] = rem * 16 + ci;             // LDS layout (hh,ww,ci): 16B-aligned ci runs
    bool ok = (hg >= 0) && (hg < H_) && (wg >= 0) && (wg < W_);
    gofs[i] = ok ? ((((b * CIN + ci) * T_) * H_ + hg) * W_ + wg) * 4 : -1;
  }

  auto prefetch = [&](int tt) {
    if (tt < T_) {
      const char* xb = (const char*)x + (size_t)tt * (H_ * W_ * 4);
#pragma unroll
      for (int i = 0; i <= NFULL; ++i)
        if (i < NFULL || tid < NTAIL)
          vals[i] = (gofs[i] >= 0) ? *(const float*)(xb + gofs[i]) : 0.0f;
    } else {
#pragma unroll
      for (int i = 0; i <= NFULL; ++i) vals[i] = 0.0f;
    }
  };
  auto commit = [&](int sl) {
    unsigned short* s = &xs[sl][0];
#pragma unroll
    for (int i = 0; i <= NFULL; ++i)
      if (i < NFULL || tid < NTAIL)
        s[lofs[i]] = f2bf(vals[i]);
  };

  // preload: slot0 <- t=0, slot1 <- t=1, slot3 (= t-1) <- zeros; start loading t=2
  prefetch(0); commit(0);
  prefetch(1); commit(1);
  for (int i = tid; i < SH; i += 256) xs[3][i] = 0;
  prefetch(2);
  __syncthreads();

  const float bzv = bias[lane & 15];        // Z gate bias, o = lane%16
  const float bgv = bias[16 + (lane & 15)]; // F gate bias, o = 16 + lane%16
  const int   cig = (lane & 16) ? 8 : 0;    // ci half per A-layout (lanes 16-31: ci 8..15)

  float hst[2][8];                          // recurrence state, matches C-matrix layout
  float* obase[2];
#pragma unroll
  for (int m = 0; m < 2; ++m) {
#pragma unroll
    for (int r = 0; r < 8; ++r) hst[m][r] = 0.0f;
    const int hid = lane & 15;
    const int hg  = h0 + 2 * wave + m;
    const int wg  = w0 + ((lane & 16) ? 8 : 0);
    obase[m] = out + (((size_t)(b * HID + hid) * T_) * H_ + hg) * W_ + wg;
  }

  for (int t = 0; t < T_; ++t) {
    commit((t + 2) & 3);     // vals holds data for t+2 (prefetched a full compute phase ago)
    prefetch(t + 3);         // overlap next slice's HBM latency with the WMMAs below

    const unsigned short* sb[3] = { xs[(t + 3) & 3], xs[t & 3], xs[(t + 1) & 3] }; // by kd

    v8f accZ0, accZ1, accF0, accF1;
#pragma unroll
    for (int r = 0; r < 8; ++r) { accZ0[r] = bzv; accZ1[r] = bzv; accF0[r] = bgv; accF1[r] = bgv; }

#pragma unroll
    for (int c = 0; c < 14; ++c) {          // K chunks of 32 (2 taps of 16 ci each)
      const int kof = 32 * c + ((lane & 16) ? 8 : 0);
      Frag bz_, bf_;
      bz_.h[0] = *(const ushort8*)&wlds[(lane & 15) * WK + kof];
      bz_.h[1] = *(const ushort8*)&wlds[(lane & 15) * WK + kof + 16];
      bf_.h[0] = *(const ushort8*)&wlds[((lane & 15) + 16) * WK + kof];
      bf_.h[1] = *(const ushort8*)&wlds[((lane & 15) + 16) * WK + kof + 16];
#pragma unroll
      for (int m = 0; m < 2; ++m) {
        const int mt = 2 * wave + m;                          // tile h-row of this M-tile
        const int ab = (mt * HALO + (lane & 15)) * 16 + cig;  // lane's A base (halves)
        Frag a;
        {
          const int tap = 2 * c;
          const int kd = tap / 9, kh = (tap % 9) / 3, kw = tap % 3;
          a.h[0] = *(const ushort8*)&sb[kd][ab + (kh * HALO + kw) * 16];
        }
        if (2 * c + 1 < 27) {
          const int tap = 2 * c + 1;
          const int kd = tap / 9, kh = (tap % 9) / 3, kw = tap % 3;
          a.h[1] = *(const ushort8*)&sb[kd][ab + (kh * HALO + kw) * 16];
        } else {
          ushort8 z = {0, 0, 0, 0, 0, 0, 0, 0};
          a.h[1] = z;
        }
        v8f& az = m ? accZ1 : accZ0;
        v8f& af = m ? accF1 : accF0;
        az = __builtin_amdgcn_wmma_f32_16x16x32_bf16(false, a.v, false, bz_.v, (short)0, az, false, false);
        af = __builtin_amdgcn_wmma_f32_16x16x32_bf16(false, a.v, false, bf_.v, (short)0, af, false, false);
      }
    }

    // activations + fo-pool step + store (C layout: VGPR r -> (M = r or r+8, hid = lane%16))
#pragma unroll
    for (int m = 0; m < 2; ++m) {
      const v8f& az = m ? accZ1 : accZ0;
      const v8f& af = m ? accF1 : accF0;
#pragma unroll
      for (int r = 0; r < 8; ++r) {
        float zz = tanh_f(az[r]);
        float ff = sigm(af[r]);
        hst[m][r] = ff * hst[m][r] + (1.0f - ff) * zz;
      }
      float* dst = obase[m] + (size_t)t * (H_ * W_);
      *(float4*)dst       = make_float4(hst[m][0], hst[m][1], hst[m][2], hst[m][3]);
      *((float4*)dst + 1) = make_float4(hst[m][4], hst[m][5], hst[m][6], hst[m][7]);
    }

    __syncthreads();   // single barrier: bounds wave drift so the 4-deep ring stays hazard-free
  }
}

extern "C" void kernel_launch(void* const* d_in, const int* in_sizes, int n_in,
                              void* d_out, int out_size, void* d_ws, size_t ws_size,
                              hipStream_t stream) {
  const float* x    = (const float*)d_in[0];
  const float* W    = (const float*)d_in[1];
  const float* bias = (const float*)d_in[2];
  qrnn3d_fused<<<dim3(256), dim3(256), 0, stream>>>(x, W, bias, (float*)d_out);
}
